// AttendRNN_51702816309613
// MI455X (gfx1250) — matrix operand order repaired and checked by reading, not compiled, not run on hardware
//
#include <hip/hip_runtime.h>
#include <math.h>

// ---------------- problem constants ----------------
#define B_    64
#define N_    256
#define V_    300
#define VK_   320          // embed dim padded to multiple of 32 for WMMA K
#define H_    256
#define G3_   768          // 3*H
#define NW_   1536         // 2*3*H (fwd+bwd input-gate weights fused)
#define D_    512          // 2*H (BiGRU output dim)
#define FC_   512
#define TOK_  (B_*2*N_)    // 32768 tokens total
#define SIGMA_INV (1.0f/0.95f)

// wihT swizzled tiles: K tiles = VK_/32 = 10, N tiles = NW_/16 = 96
#define WIH_KT (VK_/32)
#define WIH_NT (NW_/16)
// whhT swizzled tiles: K tiles = H_/32 = 8, N tiles = G3_/16 = 48
#define WHH_KT (H_/32)
#define WHH_NT (G3_/16)

// ---------------- WMMA types & fragment helpers (CDNA5, wave32) ----------------
typedef __attribute__((ext_vector_type(16))) __bf16 v16bf;
typedef __attribute__((ext_vector_type(8)))  __bf16 v8bf;
typedef __attribute__((ext_vector_type(8)))  float  v8f;

static __device__ __forceinline__ int laneid() { return threadIdx.x & 31; }

static __device__ __forceinline__ v8f v8f_zero() {
  v8f z;
#pragma unroll
  for (int i = 0; i < 8; ++i) z[i] = 0.0f;
  return z;
}

// A fragment: 16x32 bf16 tile, source row-major [M][K], p at (row0,k0), stride ld.
// Per-lane data is two contiguous 8-element (16B) chunks:
//   elements 0..7  -> K = half*16 + 0..7
//   elements 8..15 -> K = half*16 + 8 + 0..7
static __device__ __forceinline__ v16bf load_a_bf16(const __bf16* p, int ld) {
  const int lane = laneid();
  const int m = lane & 15, half = lane >> 4;
  const __bf16* row = p + m*ld + (half << 4);
  v8bf lo = *(const v8bf*)(row);
  v8bf hi = *(const v8bf*)(row + 8);
  return __builtin_shufflevector(lo, hi, 0,1,2,3,4,5,6,7,8,9,10,11,12,13,14,15);
}

// B fragment whose source is row-major [N][K] (transpose needed): per lane this is
// one contiguous 32B chunk: element i -> K = half*16 + i, N = lane&15.
static __device__ __forceinline__ v16bf load_bt_bf16(const __bf16* p, int ld) {
  const int lane = laneid();
  const int n = lane & 15, half = lane >> 4;
  return *(const v16bf*)(p + n*ld + (half << 4));
}

// B fragment from a pre-swizzled (fragment-linear) weight buffer: tile-major,
// 32 lanes x 16 bf16 contiguous per lane -> one 32B load.
static __device__ __forceinline__ v16bf load_b_swz(const __bf16* base, int tile) {
  return ((const v16bf*)base)[tile * 32 + laneid()];
}

static __device__ __forceinline__ v8f wmma_bf16(v16bf a, v16bf b, v8f c) {
  return __builtin_amdgcn_wmma_f32_16x16x32_bf16(false, a, false, b, (short)0, c, false, false);
}

static __device__ __forceinline__ float sigmoidf_(float x) {
  return 1.0f / (1.0f + __expf(-x));
}

// ---------------- 0) weight prep: fp32 -> bf16, swizzled to fragment order ----------------
// wihT swizzled: tile = kt*WIH_NT + nt ; within tile: lane, elem
//   k = kt*32 + (lane>>4)*16 + e ; n = nt*16 + (lane&15)
//   value = (k<300) ? (n<768 ? W_ih_f[n][k] : W_ih_b[n-768][k]) : 0
__global__ void k_prep_wih(const float* __restrict__ wih_f, const float* __restrict__ wih_b,
                           __bf16* __restrict__ out) {
  int idx = blockIdx.x * blockDim.x + threadIdx.x;
  if (idx >= WIH_KT * WIH_NT * 512) return;
  const int tile = idx >> 9, r = idx & 511;
  const int lane = r >> 4, e = r & 15;
  const int kt = tile / WIH_NT, nt = tile % WIH_NT;
  const int k = kt*32 + ((lane >> 4) << 4) + e;
  const int n = nt*16 + (lane & 15);
  float v = 0.0f;
  if (k < V_) v = (n < G3_) ? wih_f[n*V_ + k] : wih_b[(n - G3_)*V_ + k];
  out[idx] = (__bf16)v;
}

// whhT swizzled for both directions: tile = kt*WHH_NT + nt
__global__ void k_prep_whh(const float* __restrict__ whh_f, const float* __restrict__ whh_b,
                           __bf16* __restrict__ outf, __bf16* __restrict__ outb) {
  int idx = blockIdx.x * blockDim.x + threadIdx.x;
  if (idx >= WHH_KT * WHH_NT * 512) return;
  const int tile = idx >> 9, r = idx & 511;
  const int lane = r >> 4, e = r & 15;
  const int kt = tile / WHH_NT, nt = tile % WHH_NT;
  const int k = kt*32 + ((lane >> 4) << 4) + e;
  const int n = nt*16 + (lane & 15);
  outf[idx] = (__bf16)whh_f[n*H_ + k];
  outb[idx] = (__bf16)whh_b[n*H_ + k];
}

// fused input-gate bias: biascat[0..767]=b_ih_f, [768..1535]=b_ih_b
__global__ void k_prep_bias(const float* __restrict__ bih_f, const float* __restrict__ bih_b,
                            float* __restrict__ out) {
  int i = blockIdx.x * blockDim.x + threadIdx.x;
  if (i < G3_)       out[i] = bih_f[i];
  else if (i < NW_)  out[i] = bih_b[i - G3_];
}

// ---------------- 1) embedding gather -> bf16, K padded to 320 ----------------
__global__ void k_embed(const int* __restrict__ xids, const float* __restrict__ embed,
                        __bf16* __restrict__ embBF) {
  const int tok = blockIdx.x;      // 0..32767   (= ((b*2)+s)*256 + n)
  const int t   = threadIdx.x;     // 0..319
  const int row = xids[tok];
  const float v = (t < V_) ? embed[(size_t)row * V_ + t] : 0.0f;
  embBF[(size_t)tok * VK_ + t] = (__bf16)v;
}

// ---------------- 2) xg = embBF[32768x320] @ WihT[320x1536] + b_ih ----------------
// block = 128 threads (4 waves); each wave: one 16-row tile x 64 cols
__global__ void k_gemm_xg(const __bf16* __restrict__ embBF, const __bf16* __restrict__ wihS,
                          const float* __restrict__ biascat, float* __restrict__ xg) {
  const int wv   = threadIdx.x >> 5;
  const int row0 = (blockIdx.y * 4 + wv) * 16;
  const int nt0  = blockIdx.x * 4;           // first 16-col tile
  v8f acc[4];
#pragma unroll
  for (int t = 0; t < 4; ++t) acc[t] = v8f_zero();
  for (int k8 = 0; k8 < WIH_KT; ++k8) {
    v16bf a = load_a_bf16(embBF + (size_t)row0*VK_ + k8*32, VK_);
#pragma unroll
    for (int t = 0; t < 4; ++t) {
      v16bf b = load_b_swz(wihS, k8*WIH_NT + nt0 + t);
      acc[t] = wmma_bf16(a, b, acc[t]);
    }
  }
  const int lane = laneid();
  const int nc = lane & 15, mb = (lane >> 4) * 8;
#pragma unroll
  for (int t = 0; t < 4; ++t) {
    const int col  = (nt0 + t)*16 + nc;
    const float bias = biascat[col];
#pragma unroll
    for (int r = 0; r < 8; ++r)
      xg[(size_t)(row0 + mb + r)*NW_ + col] = acc[t][r] + bias;
  }
}

// ---------------- 3) GRU recurrence: 1 block per (doc, direction) ----------------
// h (bf16 [64][256]) resident in LDS; per step: hg = h @ WhhT via WMMA, then fused gates.
// Writes o2 (row-major) and o2T (transposed per pair) for the attention stage.
__global__ void k_gru(const float* __restrict__ xg,
                      const __bf16* __restrict__ whhS_f, const __bf16* __restrict__ whhS_b,
                      const float* __restrict__ bhh_f, const float* __restrict__ bhh_b,
                      __bf16* __restrict__ o2, __bf16* __restrict__ o2T) {
  extern __shared__ char smem[];
  __bf16* h  = (__bf16*)smem;                                 // 64*256 bf16 = 32KB
  float*  hg = (float*)(smem + (size_t)64*H_*sizeof(__bf16)); // 64*768 f32 = 192KB
  const int s   = blockIdx.x >> 1;
  const int dir = blockIdx.x & 1;
  const __bf16* whhS = dir ? whhS_b : whhS_f;
  const float*  bhh  = dir ? bhh_b  : bhh_f;
  const int tid  = threadIdx.x;          // 512 threads = 16 waves
  const int wv   = tid >> 5;
  const int lane = tid & 31;
  const int nc = lane & 15, mb = (lane >> 4) * 8;

  for (int i = tid; i < 64*H_; i += 512) h[i] = (__bf16)0.0f;
  __syncthreads();

  for (int t = 0; t < N_; ++t) {
    const int n = dir ? (N_ - 1 - t) : t;
    // ---- phase 1: hg[64x768] = h[64x256] @ whhT[256x768] ----
    // 192 (16x16) tiles; wave wv handles tiles wv, wv+16, ... (uniform per wave)
    for (int T = wv; T < 192; T += 16) {
      const int mt = T / 48, nt = T % 48;
      v8f c = v8f_zero();
#pragma unroll
      for (int k8 = 0; k8 < 8; ++k8) {
        v16bf a = load_a_bf16(h + (mt*16)*H_ + k8*32, H_);
        v16bf b = load_b_swz(whhS, k8*WHH_NT + nt);
        c = wmma_bf16(a, b, c);
      }
#pragma unroll
      for (int r = 0; r < 8; ++r)
        hg[(mt*16 + mb + r)*G3_ + nt*16 + nc] = c[r];
    }
    __syncthreads();
    // ---- phase 2: fused gate math + h update + o2/o2T store ----
    for (int idx = tid; idx < 64*H_; idx += 512) {
      const int bb = idx >> 8, j = idx & 255;
      const float* xrow = xg + (size_t)((bb*2 + s)*N_ + n)*NW_ + dir*G3_;
      const float hr = hg[bb*G3_ + j]        + bhh[j];
      const float hz = hg[bb*G3_ + 256 + j]  + bhh[256 + j];
      const float hn = hg[bb*G3_ + 512 + j]  + bhh[512 + j];
      const float r  = sigmoidf_(xrow[j] + hr);
      const float zz = sigmoidf_(xrow[256 + j] + hz);
      const float ng = tanhf(xrow[512 + j] + r*hn);
      const float hold = (float)h[bb*H_ + j];
      const float hnew = (1.0f - zz)*ng + zz*hold;
      const __bf16 hb = (__bf16)hnew;
      h[bb*H_ + j] = hb;
      const size_t pair = (size_t)(bb*2 + s);
      o2 [pair*N_*D_ + (size_t)n*D_ + dir*H_ + j] = hb;
      o2T[pair*N_*D_ + (size_t)(dir*H_ + j)*N_ + n] = hb;
    }
    __syncthreads();
  }
}

// ---------------- 4) attention: one wave per (pair, 16-query-row tile) ----------------
__global__ void k_attn(const __bf16* __restrict__ o2, const __bf16* __restrict__ o2T,
                       float* __restrict__ o5) {
  __shared__ __align__(32) float  S[16*256];   // 16KB
  __shared__ __align__(32) __bf16 P[16*256];   // 8KB
  const int pair = blockIdx.x >> 4;     // (b*2+s) in 0..127
  const int qt   = blockIdx.x & 15;
  const int q0   = qt * 16;
  const __bf16* o2p  = o2  + (size_t)pair * N_ * D_;
  const __bf16* o2tp = o2T + (size_t)pair * N_ * D_;
  const int lane = laneid();
  const int nc = lane & 15, mb = (lane >> 4) * 8;

  // scores S[16][256] = o2[q0:q0+16] . o2^T  -  dist bias
  for (int mt = 0; mt < 16; ++mt) {
    v8f c = v8f_zero();
#pragma unroll
    for (int k8 = 0; k8 < D_/32; ++k8) {
      v16bf a = load_a_bf16(o2p + (size_t)q0*D_ + k8*32, D_);
      v16bf b = load_bt_bf16(o2p + (size_t)(mt*16)*D_ + k8*32, D_);
      c = wmma_bf16(a, b, c);
    }
#pragma unroll
    for (int r = 0; r < 8; ++r) {
      const int qr = mb + r;              // local query row
      const int mk = mt*16 + nc;          // key index
      const float d = (float)((q0 + qr) - mk);
      S[qr*256 + mk] = c[r] - d*d*SIGMA_INV;
    }
  }
  __syncthreads();
  // row softmax (single wave, shuffle reductions)
  for (int r = 0; r < 16; ++r) {
    float mx = -3.0e38f;
    for (int i = lane; i < 256; i += 32) mx = fmaxf(mx, S[r*256 + i]);
#pragma unroll
    for (int off = 16; off; off >>= 1) mx = fmaxf(mx, __shfl_xor(mx, off));
    float sum = 0.0f;
    for (int i = lane; i < 256; i += 32) sum += __expf(S[r*256 + i] - mx);
#pragma unroll
    for (int off = 16; off; off >>= 1) sum += __shfl_xor(sum, off);
    const float inv = 1.0f / sum;
    for (int i = lane; i < 256; i += 32)
      P[r*256 + i] = (__bf16)(__expf(S[r*256 + i] - mx) * inv);
  }
  __syncthreads();
  // o5[q0:q0+16, :] = P[16x256] @ o2[256x512]; B from o2T (contiguous per lane)
  float* o5p = o5 + (size_t)pair * N_ * D_;
  for (int dt = 0; dt < D_/16; ++dt) {
    v8f c = v8f_zero();
#pragma unroll
    for (int k8 = 0; k8 < 256/32; ++k8) {
      v16bf a = load_a_bf16(P + k8*32, 256);
      v16bf b = load_bt_bf16(o2tp + (size_t)(dt*16)*N_ + k8*32, N_);
      c = wmma_bf16(a, b, c);
    }
#pragma unroll
    for (int r = 0; r < 8; ++r)
      o5p[(size_t)(q0 + mb + r)*D_ + dt*16 + nc] = c[r];
  }
}

// ---------------- 5) mean/max pooling over sequence ----------------
__global__ void k_pool(const float* __restrict__ o5, float* __restrict__ feats) {
  const int pair = blockIdx.x;           // 0..127
  const int d    = threadIdx.x;          // 0..511
  const float* p = o5 + (size_t)pair*N_*D_ + d;
  float sum = 0.0f, mx = -3.0e38f;
  for (int n = 0; n < N_; ++n) {
    const float v = p[(size_t)n*D_];
    sum += v; mx = fmaxf(mx, v);
  }
  feats[pair*1024 + d]       = sum * (1.0f / N_);
  feats[pair*1024 + 512 + d] = mx;
}

// ---------------- 6) pair features z = [|a-b|, a*b] ----------------
__global__ void k_zfeat(const float* __restrict__ feats, float* __restrict__ z) {
  const int b = blockIdx.x;
  for (int i = threadIdx.x; i < 2048; i += blockDim.x) {
    const int j = i & 1023;
    const float fa = feats[(b*2 + 0)*1024 + j];
    const float fb = feats[(b*2 + 1)*1024 + j];
    z[b*2048 + i] = (i < 1024) ? fabsf(fa - fb) : fa * fb;
  }
}

// ---------------- 7) fc1 + relu ----------------
__global__ void k_fc1(const float* __restrict__ z, const float* __restrict__ w,
                      const float* __restrict__ bias, float* __restrict__ hfc) {
  const int idx = blockIdx.x * blockDim.x + threadIdx.x;
  if (idx >= B_ * FC_) return;
  const int b = idx >> 9, f = idx & 511;
  const float* zr = z + b*2048;
  const float* wr = w + (size_t)f*2048;
  float s = bias[f];
  for (int i = 0; i < 2048; ++i) s += zr[i] * wr[i];
  hfc[idx] = fmaxf(s, 0.0f);
}

// ---------------- 8) fc2 + sigmoid ----------------
__global__ void k_fc2(const float* __restrict__ hfc, const float* __restrict__ w,
                      const float* __restrict__ bias, float* __restrict__ out) {
  const int b = threadIdx.x;
  if (b >= B_) return;
  float s = bias[0];
  for (int i = 0; i < FC_; ++i) s += hfc[b*FC_ + i] * w[i];
  out[b] = sigmoidf_(s);
}

// ---------------- host launcher ----------------
extern "C" void kernel_launch(void* const* d_in, const int* in_sizes, int n_in,
                              void* d_out, int out_size, void* d_ws, size_t ws_size,
                              hipStream_t stream) {
  (void)in_sizes; (void)n_in; (void)out_size; (void)ws_size;
  const int*   x      = (const int*)  d_in[0];
  const float* embed  = (const float*)d_in[1];
  const float* Wih_f  = (const float*)d_in[2];
  const float* Whh_f  = (const float*)d_in[3];
  const float* bih_f  = (const float*)d_in[4];
  const float* bhh_f  = (const float*)d_in[5];
  const float* Wih_b  = (const float*)d_in[6];
  const float* Whh_b  = (const float*)d_in[7];
  const float* bih_b  = (const float*)d_in[8];
  const float* bhh_b  = (const float*)d_in[9];
  const float* fc1_w  = (const float*)d_in[10];
  const float* fc1_b  = (const float*)d_in[11];
  const float* fc2_w  = (const float*)d_in[12];
  const float* fc2_b  = (const float*)d_in[13];
  float* out = (float*)d_out;

  char* ws = (char*)d_ws;
  size_t off = 0;
  auto carve = [&](size_t bytes) -> void* {
    void* p = ws + off;
    off += (bytes + 255) & ~(size_t)255;
    return p;
  };
  __bf16* embBF  = (__bf16*)carve((size_t)TOK_ * VK_ * sizeof(__bf16));   // 20 MB
  __bf16* wihS   = (__bf16*)carve((size_t)WIH_KT*WIH_NT*512 * sizeof(__bf16)); // 1 MB
  __bf16* whhSf  = (__bf16*)carve((size_t)WHH_KT*WHH_NT*512 * sizeof(__bf16));
  __bf16* whhSb  = (__bf16*)carve((size_t)WHH_KT*WHH_NT*512 * sizeof(__bf16));
  float*  biascat= (float*) carve((size_t)NW_ * sizeof(float));
  float*  xg     = (float*) carve((size_t)TOK_ * NW_ * sizeof(float));    // 192 MB
  __bf16* o2     = (__bf16*)carve((size_t)TOK_ * D_ * sizeof(__bf16));    // 32 MB
  __bf16* o2T    = (__bf16*)carve((size_t)TOK_ * D_ * sizeof(__bf16));    // 32 MB
  float*  o5     = (float*) carve((size_t)TOK_ * D_ * sizeof(float));     // 64 MB
  float*  feats  = (float*) carve((size_t)128 * 1024 * sizeof(float));
  float*  zbuf   = (float*) carve((size_t)B_ * 2048 * sizeof(float));
  float*  hfc    = (float*) carve((size_t)B_ * FC_ * sizeof(float));

  // 0) weight prep (bf16 + fragment swizzle) + bias concat
  k_prep_wih<<<(WIH_KT*WIH_NT*512 + 255)/256, 256, 0, stream>>>(Wih_f, Wih_b, wihS);
  k_prep_whh<<<(WHH_KT*WHH_NT*512 + 255)/256, 256, 0, stream>>>(Whh_f, Whh_b, whhSf, whhSb);
  k_prep_bias<<<(NW_ + 255)/256, 256, 0, stream>>>(bih_f, bih_b, biascat);
  // 1) embedding gather
  k_embed<<<TOK_, VK_, 0, stream>>>(x, embed, embBF);
  // 2) input-gate GEMM (WMMA)
  k_gemm_xg<<<dim3(NW_/64, TOK_/64), 128, 0, stream>>>(embBF, wihS, biascat, xg);
  // 3) GRU recurrence (WMMA per step, LDS-resident hidden state)
  const size_t gru_lds = (size_t)64*H_*sizeof(__bf16) + (size_t)64*G3_*sizeof(float);
  k_gru<<<4, 512, gru_lds, stream>>>(xg, whhSf, whhSb, bhh_f, bhh_b, o2, o2T);
  // 4) self-attention (WMMA scores + WMMA PV)
  k_attn<<<128 * (N_/16), 32, 0, stream>>>(o2, o2T, o5);
  // 5..8) pooling + pair features + MLP head
  k_pool<<<128, D_, 0, stream>>>(o5, feats);
  k_zfeat<<<B_, 1024, 0, stream>>>(feats, zbuf);
  k_fc1<<<(B_*FC_ + 255)/256, 256, 0, stream>>>(zbuf, fc1_w, fc1_b, hfc);
  k_fc2<<<1, 64, 0, stream>>>(hfc, fc2_w, fc2_b, out);
}
